// ImprovedGraphTransformer_43688407335637
// MI455X (gfx1250) — matrix operand order repaired and checked by reading, not compiled, not run on hardware
//
#include <hip/hip_runtime.h>
#include <math.h>

typedef __attribute__((ext_vector_type(16))) _Float16 v16h;
typedef __attribute__((ext_vector_type(8)))  _Float16 v8h;
typedef __attribute__((ext_vector_type(8)))  float    v8f;

#define WMMA_F16(a, b, c) \
  __builtin_amdgcn_wmma_f32_16x16x32_f16(false, (a), false, (b), (short)0, (c), false, false)

// ---------------------------------------------------------------------------
// Fragment loaders (wave32, 16x16x32 f16 WMMA)
// A layout: lane grp g (lane>>4), row = lane&15;
//   halfs j -> K = kb + g*16 + j
// B layout (sparse-B striping): col = lane&15; halfs j -> K = kb + g*16 + j
// C/D layout: reg r holds M = r + 8*g, N = lane&15
// ---------------------------------------------------------------------------
__device__ __forceinline__ v16h fragA_f32(const float* p, int grp) {
  const float* p0 = p + grp * 16;
  v16h f;
#pragma unroll
  for (int j = 0; j < 8; ++j) f[j] = (_Float16)p0[j];
#pragma unroll
  for (int j = 0; j < 8; ++j) f[8 + j] = (_Float16)p0[8 + j];
  return f;
}

__device__ __forceinline__ v16h fragB_contig_f32(const float* p, int grp) {
  const float* q = p + grp * 16;
  v16h f;
#pragma unroll
  for (int j = 0; j < 16; ++j) f[j] = (_Float16)q[j];
  return f;
}

__device__ __forceinline__ v16h fragB_strided_f32(const float* W, int ldw, int col,
                                                  int kb, int grp) {
  const float* p = W + (size_t)(kb + grp * 16) * ldw + col;
  v16h f;
#pragma unroll
  for (int j = 0; j < 16; ++j) f[j] = (_Float16)p[(size_t)j * ldw];
  return f;
}

// LDS holds W transposed as halfs: lds[col*256 + k]  -> 32B contiguous per lane
__device__ __forceinline__ v16h fragB_lds(const _Float16* lds, int col, int kb, int grp) {
  const _Float16* p = lds + (size_t)col * 256 + kb + grp * 16;
  v8h lo = *(const v8h*)p;
  v8h hi = *(const v8h*)(p + 8);
  return __builtin_shufflevector(lo, hi, 0, 1, 2, 3, 4, 5, 6, 7, 8, 9, 10, 11, 12, 13, 14, 15);
}

// ---------------------------------------------------------------------------
// Generic GEMM: Y[M,N] = X[M,K] @ W[K,N] (+ bias[N]); one wave per 16x16 tile
// ---------------------------------------------------------------------------
__global__ void wmma_gemm_xw(const float* __restrict__ X, const float* __restrict__ W,
                             const float* __restrict__ bias, float* __restrict__ Y,
                             int M, int K, int Nw) {
  int lane = threadIdx.x & 31, grp = lane >> 4, ln = lane & 15;
  int w = blockIdx.x * (blockDim.x >> 5) + (threadIdx.x >> 5);
  int ntiles = Nw >> 4;
  int tiles = (M >> 4) * ntiles;
  if (w >= tiles) return;
  int mt = w / ntiles, nt = w % ntiles;
  int arow = mt * 16 + ln, col = nt * 16 + ln;
  v8f acc = {};
  int kch = K >> 5;
  for (int kc = 0; kc < kch; ++kc) {
    v16h a = fragA_f32(X + (size_t)arow * K + kc * 32, grp);
    v16h b = fragB_strided_f32(W, Nw, col, kc * 32, grp);
    acc = WMMA_F16(a, b, acc);
  }
  float bi = bias ? bias[col] : 0.0f;
#pragma unroll
  for (int r = 0; r < 8; ++r)
    Y[(size_t)(mt * 16 + r + 8 * grp) * Nw + col] = acc[r] + bi;
}

// ---------------------------------------------------------------------------
// scores[b,h,n,m] = Q[b,n,h,:] . K[b,m,h,:] / sqrt(32); one wave per 16x16 tile
// ---------------------------------------------------------------------------
__global__ void scores_kernel(const float* __restrict__ Q, const float* __restrict__ Km,
                              float* __restrict__ S) {
  int lane = threadIdx.x & 31, grp = lane >> 4, ln = lane & 15;
  int w = blockIdx.x * (blockDim.x >> 5) + (threadIdx.x >> 5);  // < 4096
  int bh = w >> 8;
  int t = w & 255, mt = t >> 4, nt = t & 15;
  int b = bh >> 3, h = bh & 7;
  int nrow = mt * 16 + ln;  // query index
  int mrow = nt * 16 + ln;  // key index
  v16h a = fragA_f32(Q + ((size_t)(b * 256 + nrow)) * 256 + h * 32, grp);
  v16h bf = fragB_contig_f32(Km + ((size_t)(b * 256 + mrow)) * 256 + h * 32, grp);
  v8f acc = {};
  acc = WMMA_F16(a, bf, acc);
  const float sc = 0.17677669529663687f;  // 1/sqrt(32)
  size_t base = ((size_t)bh) << 16;
#pragma unroll
  for (int r = 0; r < 8; ++r)
    S[base + (size_t)(mt * 16 + r + 8 * grp) * 256 + nt * 16 + ln] = acc[r] * sc;
}

// ---------------------------------------------------------------------------
// Fused edge-bias kernel: for each flattened (b,n,m) row of edge_features,
//   edge_h = E@Wep+bep ; gate = sigmoid(E@Weg+beg) ;
//   EA[b,h,n,m] = sum_d edge_h[.,32h+d]*gate[.,32h+d]
// E read from HBM once; Wep/Weg staged transposed in LDS as f16 (256 KB,
// shared by 16 waves per block for occupancy: 4 waves/SIMD on the WGP).
// One wave per 16-row M-tile; 256 WMMAs per wave.
// ---------------------------------------------------------------------------
__global__ void edge_attn_kernel(const float* __restrict__ E,
                                 const float* __restrict__ Wep, const float* __restrict__ bep,
                                 const float* __restrict__ Weg, const float* __restrict__ beg,
                                 float* __restrict__ EA) {
  extern __shared__ _Float16 lds[];   // [2][256 cols][256 k]
  _Float16* lp = lds;
  _Float16* lg = lds + 65536;
  int tid = threadIdx.x;              // blockDim = 512 (16 waves)
  for (int i = tid; i < 65536; i += 512) {
    int k = i >> 8, n = i & 255;
    lp[n * 256 + k] = (_Float16)Wep[i];
    lg[n * 256 + k] = (_Float16)Weg[i];
  }
  __syncthreads();

  int lane = tid & 31, grp = lane >> 4, ln = lane & 15;
  int mtile = blockIdx.x * 16 + (tid >> 5);  // 8192 total
  int R = mtile * 16;                        // flattened (b,n,m) row base
  int b = R >> 16, rem = R & 65535, n = rem >> 8, m0 = rem & 255;

  // A-panel: 16 rows x 256 K held in registers as 8 fragments (64 VGPRs)
  v16h A[8];
#pragma unroll
  for (int kc = 0; kc < 8; ++kc)
    A[kc] = fragA_f32(E + (size_t)(R + ln) * 256 + kc * 32, grp);

  for (int h = 0; h < 8; ++h) {
    float hs[8];
#pragma unroll
    for (int r = 0; r < 8; ++r) hs[r] = 0.0f;
#pragma unroll
    for (int p = 0; p < 2; ++p) {      // two 16-col tiles per head (D = 32)
      int col = (h * 2 + p) * 16 + ln;
      v8f ah = {}, ag = {};
#pragma unroll
      for (int kc = 0; kc < 8; ++kc) {
        v16h bh_ = fragB_lds(lp, col, kc * 32, grp);
        ah = WMMA_F16(A[kc], bh_, ah);
        v16h bg_ = fragB_lds(lg, col, kc * 32, grp);
        ag = WMMA_F16(A[kc], bg_, ag);
      }
      float bpv = bep[col], bgv = beg[col];
#pragma unroll
      for (int r = 0; r < 8; ++r) {
        float eh = ah[r] + bpv;
        float gg = ag[r] + bgv;
        hs[r] += eh * (1.0f / (1.0f + __expf(-gg)));
      }
    }
    // reduce over the 16 columns (lanes within each half-wave)
#pragma unroll
    for (int r = 0; r < 8; ++r) {
      hs[r] += __shfl_xor(hs[r], 1, 32);
      hs[r] += __shfl_xor(hs[r], 2, 32);
      hs[r] += __shfl_xor(hs[r], 4, 32);
      hs[r] += __shfl_xor(hs[r], 8, 32);
    }
    if (ln == 0) {
      size_t base = (((size_t)(b * 8 + h)) << 16) + (size_t)n * 256 + m0;
#pragma unroll
      for (int r = 0; r < 8; ++r) EA[base + r + 8 * grp] = hs[r];
    }
  }
}

// ---------------------------------------------------------------------------
// Masked softmax over m, in place on S; one wave per (b,h,n) row
// ---------------------------------------------------------------------------
__global__ void softmax_kernel(float* __restrict__ S, const float* __restrict__ EA,
                               const int* __restrict__ adj) {
  int lane = threadIdx.x & 31;
  int row = blockIdx.x * (blockDim.x >> 5) + (threadIdx.x >> 5);  // < 4096
  int b = row >> 11;          // H*N = 2048 rows per batch
  int n = row & 255;
  size_t base = (size_t)row * 256;
  const int* arow = adj + ((size_t)(b * 256 + n)) * 256;
  float v[8];
  float mx = -3.0e38f;
#pragma unroll
  for (int j = 0; j < 8; ++j) {
    int m = j * 32 + lane;
    float x = S[base + m] + EA[base + m];
    if (arow[m] == 0) x = -1.0e9f;
    v[j] = x;
    mx = fmaxf(mx, x);
  }
  for (int off = 16; off > 0; off >>= 1) mx = fmaxf(mx, __shfl_xor(mx, off, 32));
  float sum = 0.0f;
#pragma unroll
  for (int j = 0; j < 8; ++j) { v[j] = __expf(v[j] - mx); sum += v[j]; }
  for (int off = 16; off > 0; off >>= 1) sum += __shfl_xor(sum, off, 32);
  float inv = 1.0f / sum;
#pragma unroll
  for (int j = 0; j < 8; ++j) S[base + j * 32 + lane] = v[j] * inv;
}

__global__ void attn_mean_kernel(const float* __restrict__ attn, float* __restrict__ am) {
  int gid = blockIdx.x * 256 + threadIdx.x;  // < 131072
  int b = gid >> 16, nm = gid & 65535;
  float s = 0.0f;
#pragma unroll
  for (int h = 0; h < 8; ++h) s += attn[(((size_t)(b * 8 + h)) << 16) + nm];
  am[gid] = s * 0.125f;
}

// ---------------------------------------------------------------------------
// attn_out[b,n,32h+d] = sum_m attn[b,h,n,m] * V[b,m,32h+d]
// ---------------------------------------------------------------------------
__global__ void attn_v_kernel(const float* __restrict__ Attn, const float* __restrict__ V,
                              float* __restrict__ AO) {
  int lane = threadIdx.x & 31, grp = lane >> 4, ln = lane & 15;
  int w = blockIdx.x * (blockDim.x >> 5) + (threadIdx.x >> 5);  // < 512
  int b = w >> 8, rem = w & 255;
  int h = rem >> 5, rem2 = rem & 31;
  int mt = rem2 >> 1, nt = rem2 & 1;
  int nrow = mt * 16 + ln;
  int col = nt * 16 + ln;  // d within head
  const float* Ab = Attn + (((size_t)(b * 8 + h)) << 16);
  const float* Vb = V + ((size_t)b << 16);
  v8f acc = {};
  for (int kc = 0; kc < 8; ++kc) {
    v16h a = fragA_f32(Ab + (size_t)nrow * 256 + kc * 32, grp);
    v16h bf = fragB_strided_f32(Vb, 256, h * 32 + col, kc * 32, grp);
    acc = WMMA_F16(a, bf, acc);
  }
#pragma unroll
  for (int r = 0; r < 8; ++r)
    AO[((size_t)(b * 256 + mt * 16 + r + 8 * grp)) * 256 + h * 32 + col] = acc[r];
}

// ---------------------------------------------------------------------------
// LayerNorm helpers / kernels
// ---------------------------------------------------------------------------
__device__ __forceinline__ float block_sum_256(float v, float* red, int tid) {
  red[tid] = v;
  __syncthreads();
  for (int s = 128; s > 0; s >>= 1) {
    if (tid < s) red[tid] += red[tid + s];
    __syncthreads();
  }
  float r = red[0];
  __syncthreads();
  return r;
}

__device__ __forceinline__ float block_sum_64(float v, float* red, int tid) {
  red[tid] = v;
  __syncthreads();
  for (int s = 32; s > 0; s >>= 1) {
    if (tid < s) red[tid] += red[tid + s];
    __syncthreads();
  }
  float r = red[0];
  __syncthreads();
  return r;
}

__global__ void ln_node_kernel(const float* __restrict__ resid, const float* __restrict__ y,
                               const float* __restrict__ g, const float* __restrict__ bta,
                               float* __restrict__ out) {
  __shared__ float red[256];
  int row = blockIdx.x, c = threadIdx.x;
  float x = resid[(size_t)row * 256 + c] + y[(size_t)row * 256 + c];
  float mu = block_sum_256(x, red, c) * (1.0f / 256.0f);
  float d = x - mu;
  float var = block_sum_256(d * d, red, c) * (1.0f / 256.0f);
  out[(size_t)row * 256 + c] = d * rsqrtf(var + 1e-5f) * g[c] + bta[c];
}

// edge_out = LN(edge + attn_mean*0.5*(P[n]+P[m]) + beo), P = h_out @ Weo
// float4 path: 64 threads per (b,n,m) row -> global_load/store_b128, full
// 512B/wave/instruction to saturate HBM on the 270 MB stream.
__global__ void edge_out_kernel(const float4* __restrict__ E4, const float* __restrict__ AM,
                                const float4* __restrict__ P4, const float4* __restrict__ beo4,
                                const float4* __restrict__ g24, const float4* __restrict__ b24,
                                float4* __restrict__ O4) {
  __shared__ float red[64];
  int row = blockIdx.x;  // flattened (b,n,m), 131072 rows
  int c = threadIdx.x;   // 64 threads, 4 channels each
  int b = row >> 16, nm = row & 65535, n = nm >> 8, m = nm & 255;
  float am = AM[row] * 0.5f;
  float4 pn = P4[(size_t)(b * 256 + n) * 64 + c];
  float4 pm = P4[(size_t)(b * 256 + m) * 64 + c];
  float4 e  = E4[(size_t)row * 64 + c];
  float4 bo = beo4[c];
  float4 x;
  x.x = e.x + am * (pn.x + pm.x) + bo.x;
  x.y = e.y + am * (pn.y + pm.y) + bo.y;
  x.z = e.z + am * (pn.z + pm.z) + bo.z;
  x.w = e.w + am * (pn.w + pm.w) + bo.w;
  float mu = block_sum_64(x.x + x.y + x.z + x.w, red, c) * (1.0f / 256.0f);
  float4 d;
  d.x = x.x - mu; d.y = x.y - mu; d.z = x.z - mu; d.w = x.w - mu;
  float var = block_sum_64(d.x * d.x + d.y * d.y + d.z * d.z + d.w * d.w, red, c)
              * (1.0f / 256.0f);
  float inv = rsqrtf(var + 1e-5f);
  float4 g = g24[c], bb = b24[c], o;
  o.x = d.x * inv * g.x + bb.x;
  o.y = d.y * inv * g.y + bb.y;
  o.z = d.z * inv * g.z + bb.z;
  o.w = d.w * inv * g.w + bb.w;
  O4[(size_t)row * 64 + c] = o;
}

// ---------------------------------------------------------------------------
extern "C" void kernel_launch(void* const* d_in, const int* in_sizes, int n_in,
                              void* d_out, int out_size, void* d_ws, size_t ws_size,
                              hipStream_t stream) {
  const float* node = (const float*)d_in[0];
  const float* edge = (const float*)d_in[1];
  const int*   adj  = (const int*)d_in[2];
  const float* Wn  = (const float*)d_in[3],  *bn  = (const float*)d_in[4];
  const float* Wq  = (const float*)d_in[5],  *bq  = (const float*)d_in[6];
  const float* Wk  = (const float*)d_in[7],  *bk  = (const float*)d_in[8];
  const float* Wv  = (const float*)d_in[9],  *bv  = (const float*)d_in[10];
  const float* Wep = (const float*)d_in[11], *bep = (const float*)d_in[12];
  const float* Weg = (const float*)d_in[13], *beg = (const float*)d_in[14];
  const float* Wo  = (const float*)d_in[15], *bo  = (const float*)d_in[16];
  const float* Weo = (const float*)d_in[17], *beo = (const float*)d_in[18];
  const float* g1  = (const float*)d_in[19], *b1  = (const float*)d_in[20];
  const float* g2  = (const float*)d_in[21], *b2  = (const float*)d_in[22];

  float* ws  = (float*)d_ws;
  float* h   = ws;                 // 131072
  float* Q   = ws + 131072;        // 131072
  float* Kt  = ws + 262144;        // 131072
  float* Vt  = ws + 393216;        // 131072
  float* S   = ws + 524288;        // 1048576 (scores -> attn in place)
  float* EA  = ws + 1572864;       // 1048576
  float* AM  = ws + 2621440;       // 131072
  float* AO  = ws + 2752512;       // 131072
  float* AO2 = ws + 2883584;       // 131072
  float* P   = ws + 3014656;       // 131072

  float* hout = (float*)d_out;            // [2,256,256]
  float* eout = (float*)d_out + 131072;   // [2,256,256,256]

  // node path
  wmma_gemm_xw<<<64, 256, 0, stream>>>(node, Wn, bn, h, 512, 128, 256);
  wmma_gemm_xw<<<64, 256, 0, stream>>>(h, Wq, bq, Q, 512, 256, 256);
  wmma_gemm_xw<<<64, 256, 0, stream>>>(h, Wk, bk, Kt, 512, 256, 256);
  wmma_gemm_xw<<<64, 256, 0, stream>>>(h, Wv, bv, Vt, 512, 256, 256);
  scores_kernel<<<512, 256, 0, stream>>>(Q, Kt, S);

  // edge bias (the big one): 256 KB dynamic LDS shared by 16 waves/block
  edge_attn_kernel<<<512, 512, 262144, stream>>>(edge, Wep, bep, Weg, beg, EA);

  softmax_kernel<<<512, 256, 0, stream>>>(S, EA, adj);
  attn_mean_kernel<<<512, 256, 0, stream>>>(S, AM);
  attn_v_kernel<<<64, 256, 0, stream>>>(S, Vt, AO);
  wmma_gemm_xw<<<64, 256, 0, stream>>>(AO, Wo, bo, AO2, 512, 256, 256);
  ln_node_kernel<<<512, 256, 0, stream>>>(h, AO2, g1, b1, hout);

  // edge update, factored: P = h_out @ Weo, then elementwise + LN (float4)
  wmma_gemm_xw<<<64, 256, 0, stream>>>(hout, Weo, nullptr, P, 512, 256, 256);
  edge_out_kernel<<<131072, 64, 0, stream>>>((const float4*)edge, AM, (const float4*)P,
                                             (const float4*)beo, (const float4*)g2,
                                             (const float4*)b2, (float4*)eout);
}
